// CausalSelfAttention_62697932587259
// MI455X (gfx1250) — matrix operand order repaired and checked by reading, not compiled, not run on hardware
//
#include <hip/hip_runtime.h>

typedef __attribute__((ext_vector_type(16))) __bf16 v16bf;
typedef __attribute__((ext_vector_type(8)))  float  v8f;

struct FragBF { union { unsigned int u[8]; v16bf v; }; };

__device__ __forceinline__ unsigned short f2bf(float f) {
    unsigned int u = __float_as_uint(f);
    unsigned int r = u + 0x7FFFu + ((u >> 16) & 1u);   // round-to-nearest-even
    return (unsigned short)(r >> 16);
}
__device__ __forceinline__ unsigned int ld_u32(const unsigned short* p) {
    return *(const unsigned int*)p;
}

constexpr int cB = 4, cT = 2048, cC = 1024, cH = 16, cD = 64;
constexpr int cM = cB * cT;          // 8192 tokens
constexpr int c3C = 3 * cC;          // 3072

// A-fragment (16x32 bf16): lane m = lane&15, half = lane>>4.
// rowPtr must already point at (row m, column k0).
__device__ __forceinline__ void load_a(FragBF& a, const unsigned short* rowPtr, int half) {
#pragma unroll
    for (int j = 0; j < 8; ++j) {
        int dbase = half * 16 + 2 * j;
        a.u[j] = ld_u32(rowPtr + dbase);
    }
}
// B-fragment (32x16 bf16): lane n = lane&15, K-half = lane>>4.
// p must already point at (row n, column k0 + half*16).
__device__ __forceinline__ void load_b(FragBF& b, const unsigned short* p) {
#pragma unroll
    for (int j = 0; j < 8; ++j) b.u[j] = ld_u32(p + 2 * j);
}

// ---------------- conversion kernels ----------------
__global__ void k_cvt_bf16(const float* __restrict__ in, unsigned short* __restrict__ out, int n) {
    int i = blockIdx.x * blockDim.x + threadIdx.x;
    if (i < n) out[i] = f2bf(in[i]);
}

// W is [K][N] row-major (f32); WT is [N][K] bf16
__global__ void k_transpose_bf16(const float* __restrict__ W, unsigned short* __restrict__ WT,
                                 int K, int N) {
    int i = blockIdx.x * blockDim.x + threadIdx.x;
    if (i < K * N) {
        int n = i / K;
        int k = i - n * K;
        WT[i] = f2bf(W[(size_t)k * N + n]);
    }
}

// ---------------- QKV projection GEMM ----------------
// Each wave: 32 rows x 64 cols, double-buffered K loop (32 per step).
__global__ __launch_bounds__(256) void k_gemm_qkv(
        const unsigned short* __restrict__ xbf,   // [M][C] bf16
        const unsigned short* __restrict__ wT,    // [3C][C] bf16 (pre-transposed)
        const float* __restrict__ bias,           // [3C]
        unsigned short* __restrict__ qb,
        unsigned short* __restrict__ kb,
        unsigned short* __restrict__ vtb) {
    const int lane = threadIdx.x & 31;
    const int wave = threadIdx.x >> 5;
    const int nIdx = lane & 15;
    const int half = lane >> 4;
    const int n0   = blockIdx.x * 64;
    const int mrow = blockIdx.y * 256 + wave * 32;

    const unsigned short* aP0 = xbf + (size_t)(mrow + nIdx) * cC;
    const unsigned short* aP1 = xbf + (size_t)(mrow + 16 + nIdx) * cC;
    const unsigned short* bP[4];
#pragma unroll
    for (int nt = 0; nt < 4; ++nt)
        bP[nt] = wT + (size_t)(n0 + nt * 16 + nIdx) * cC + half * 16;

    v8f z = {0.f,0.f,0.f,0.f,0.f,0.f,0.f,0.f};
    v8f acc[2][4] = {{z,z,z,z},{z,z,z,z}};

    FragBF aC[2], bC[4], aN[2], bN[4];
    load_a(aC[0], aP0, half); load_a(aC[1], aP1, half);
#pragma unroll
    for (int nt = 0; nt < 4; ++nt) load_b(bC[nt], bP[nt]);

    for (int k0 = 0; k0 < cC; k0 += 64) {
        const int kn = k0 + 32;                 // always < cC (k0 <= cC-64)
        load_a(aN[0], aP0 + kn, half); load_a(aN[1], aP1 + kn, half);
#pragma unroll
        for (int nt = 0; nt < 4; ++nt) load_b(bN[nt], bP[nt] + kn);
#pragma unroll
        for (int nt = 0; nt < 4; ++nt) {
            acc[0][nt] = __builtin_amdgcn_wmma_f32_16x16x32_bf16(
                false, aC[0].v, false, bC[nt].v, (short)0, acc[0][nt], false, false);
            acc[1][nt] = __builtin_amdgcn_wmma_f32_16x16x32_bf16(
                false, aC[1].v, false, bC[nt].v, (short)0, acc[1][nt], false, false);
        }
        const int kc = k0 + 64;
        if (kc < cC) {
            load_a(aC[0], aP0 + kc, half); load_a(aC[1], aP1 + kc, half);
#pragma unroll
            for (int nt = 0; nt < 4; ++nt) load_b(bC[nt], bP[nt] + kc);
        }
#pragma unroll
        for (int nt = 0; nt < 4; ++nt) {
            acc[0][nt] = __builtin_amdgcn_wmma_f32_16x16x32_bf16(
                false, aN[0].v, false, bN[nt].v, (short)0, acc[0][nt], false, false);
            acc[1][nt] = __builtin_amdgcn_wmma_f32_16x16x32_bf16(
                false, aN[1].v, false, bN[nt].v, (short)0, acc[1][nt], false, false);
        }
    }
#pragma unroll
    for (int mi = 0; mi < 2; ++mi) {
#pragma unroll
        for (int nt = 0; nt < 4; ++nt) {
#pragma unroll
            for (int r = 0; r < 8; ++r) {
                int n = n0 + nt * 16 + nIdx;
                int m = mrow + mi * 16 + r + 8 * half;
                unsigned short bv = f2bf(acc[mi][nt][r] + bias[n]);
                int sel = n >> 10, c = n & 1023, head = c >> 6, d = c & 63;
                int b = m >> 11, t = m & 2047;
                if (sel == 0)
                    qb[(((size_t)b * cH + head) * cT + t) * cD + d] = bv;
                else if (sel == 1)
                    kb[(((size_t)b * cH + head) * cT + t) * cD + d] = bv;
                else
                    vtb[(((size_t)b * cH + head) * cD + d) * cT + t] = bv;
            }
        }
    }
}

// ---------------- flash attention ----------------
__global__ __launch_bounds__(256) void k_attn(
        const unsigned short* __restrict__ qb,   // [B,H,T,D]
        const unsigned short* __restrict__ kb,   // [B,H,T,D]
        const unsigned short* __restrict__ vtb,  // [B,H,D,T]
        unsigned short* __restrict__ attnbf) {   // [B,T,C]
    __shared__ unsigned short Plds[8][16][32];   // per-wave P staging, 8 KB

    const int lane = threadIdx.x & 31;
    const int wave = threadIdx.x >> 5;
    const int nIdx = lane & 15;
    const int half = lane >> 4;
    const int bh   = blockIdx.x;                 // b*H + head
    const int b    = bh >> 4, head = bh & 15;
    const int q0   = blockIdx.y * 128 + wave * 16;
    const float scale = 0.125f;                  // 1/sqrt(64)

    const unsigned short* qBase  = qb  + (size_t)bh * cT * cD;
    const unsigned short* kBase  = kb  + (size_t)bh * cT * cD;
    const unsigned short* vtBase = vtb + (size_t)bh * cD * cT;

    // Q fragments (A layout), d in [0,32) and [32,64)
    FragBF qa0, qa1;
    {
        const unsigned short* qRow = qBase + (size_t)(q0 + nIdx) * cD;
        load_a(qa0, qRow, half);
        load_a(qa1, qRow + 32, half);
    }

    v8f z = {0.f,0.f,0.f,0.f,0.f,0.f,0.f,0.f};
    v8f o[4] = {z, z, z, z};
    float mrun[8], lrun[8];
#pragma unroll
    for (int r = 0; r < 8; ++r) { mrun[r] = -3e30f; lrun[r] = 0.f; }

    const int kEnd = q0 + 16;
    for (int k0 = 0; k0 < kEnd; k0 += 32) {
        // ---- load K fragments [key subtile][d-half] and all V fragments up front
        FragBF kf[2][2], vf[4];
#pragma unroll
        for (int s = 0; s < 2; ++s) {
            const unsigned short* kr = kBase + (size_t)(k0 + s * 16 + nIdx) * cD + half * 16;
            load_b(kf[s][0], kr);
            load_b(kf[s][1], kr + 32);
        }
#pragma unroll
        for (int dt = 0; dt < 4; ++dt)
            load_b(vf[dt], vtBase + (size_t)(dt * 16 + nIdx) * cT + k0 + half * 16);

        // prefetch next key tile (gfx1250 global_prefetch path)
        if (k0 + 32 < kEnd) {
#pragma unroll
            for (int s = 0; s < 2; ++s)
                __builtin_prefetch(kBase + (size_t)(k0 + 32 + s * 16 + nIdx) * cD, 0, 3);
        }

        // ---- S = Q K^T (two 16x16 key subtiles, K-dim = 64 head dim)
        v8f s0 = z, s1 = z;
        s0 = __builtin_amdgcn_wmma_f32_16x16x32_bf16(false, qa0.v, false, kf[0][0].v, (short)0, s0, false, false);
        s0 = __builtin_amdgcn_wmma_f32_16x16x32_bf16(false, qa1.v, false, kf[0][1].v, (short)0, s0, false, false);
        s1 = __builtin_amdgcn_wmma_f32_16x16x32_bf16(false, qa0.v, false, kf[1][0].v, (short)0, s1, false, false);
        s1 = __builtin_amdgcn_wmma_f32_16x16x32_bf16(false, qa1.v, false, kf[1][1].v, (short)0, s1, false, false);

        // ---- online softmax per row (rows r+8*half live across 16 lanes of each half)
#pragma unroll
        for (int r = 0; r < 8; ++r) {
            int m  = r + 8 * half;
            int qi = q0 + m;
            float a0 = s0[r] * scale; if (k0 + nIdx      > qi) a0 = -3e30f;
            float a1 = s1[r] * scale; if (k0 + 16 + nIdx > qi) a1 = -3e30f;
            float pm = fmaxf(a0, a1);
#pragma unroll
            for (int off = 1; off < 16; off <<= 1) pm = fmaxf(pm, __shfl_xor(pm, off, 32));
            float mnew  = fmaxf(mrun[r], pm);
            float alpha = __expf(mrun[r] - mnew);
            float p0 = __expf(a0 - mnew);
            float p1 = __expf(a1 - mnew);
            float rs = p0 + p1;
#pragma unroll
            for (int off = 1; off < 16; off <<= 1) rs += __shfl_xor(rs, off, 32);
            lrun[r] = lrun[r] * alpha + rs;
            mrun[r] = mnew;
            o[0][r] *= alpha; o[1][r] *= alpha; o[2][r] *= alpha; o[3][r] *= alpha;
            Plds[wave][m][nIdx]      = f2bf(p0);
            Plds[wave][m][16 + nIdx] = f2bf(p1);
        }
        asm volatile("s_wait_dscnt 0x0" ::: "memory");

        // ---- re-fragment P (C-layout in LDS) into A-layout
        FragBF pa;
#pragma unroll
        for (int j = 0; j < 8; ++j) {
            int kbase = half * 16 + 2 * j;
            pa.u[j] = ld_u32(&Plds[wave][nIdx][kbase]);
        }
        // ---- O += P * V (V fragments already in registers)
#pragma unroll
        for (int dt = 0; dt < 4; ++dt)
            o[dt] = __builtin_amdgcn_wmma_f32_16x16x32_bf16(
                false, pa.v, false, vf[dt].v, (short)0, o[dt], false, false);
    }

    // normalize and store attn output [B,T,C] bf16
#pragma unroll
    for (int r = 0; r < 8; ++r) {
        float inv = 1.0f / lrun[r];
        int m  = r + 8 * half;
        size_t rowOff = ((size_t)b * cT + (q0 + m)) * cC + head * cD;
#pragma unroll
        for (int dt = 0; dt < 4; ++dt)
            attnbf[rowOff + dt * 16 + nIdx] = f2bf(o[dt][r] * inv);
    }
}

// ---------------- output projection GEMM ----------------
__global__ __launch_bounds__(256) void k_gemm_out(
        const unsigned short* __restrict__ attnbf,  // [M][C] bf16
        const unsigned short* __restrict__ woT,     // [C][C] bf16 (pre-transposed)
        const float* __restrict__ bo,
        float* __restrict__ out) {
    const int lane = threadIdx.x & 31;
    const int wave = threadIdx.x >> 5;
    const int nIdx = lane & 15;
    const int half = lane >> 4;
    const int n0   = blockIdx.x * 64;
    const int mrow = blockIdx.y * 256 + wave * 32;

    const unsigned short* aP0 = attnbf + (size_t)(mrow + nIdx) * cC;
    const unsigned short* aP1 = attnbf + (size_t)(mrow + 16 + nIdx) * cC;
    const unsigned short* bP[4];
#pragma unroll
    for (int nt = 0; nt < 4; ++nt)
        bP[nt] = woT + (size_t)(n0 + nt * 16 + nIdx) * cC + half * 16;

    v8f z = {0.f,0.f,0.f,0.f,0.f,0.f,0.f,0.f};
    v8f acc[2][4] = {{z,z,z,z},{z,z,z,z}};

    FragBF aC[2], bC[4], aN[2], bN[4];
    load_a(aC[0], aP0, half); load_a(aC[1], aP1, half);
#pragma unroll
    for (int nt = 0; nt < 4; ++nt) load_b(bC[nt], bP[nt]);

    for (int k0 = 0; k0 < cC; k0 += 64) {
        const int kn = k0 + 32;
        load_a(aN[0], aP0 + kn, half); load_a(aN[1], aP1 + kn, half);
#pragma unroll
        for (int nt = 0; nt < 4; ++nt) load_b(bN[nt], bP[nt] + kn);
#pragma unroll
        for (int nt = 0; nt < 4; ++nt) {
            acc[0][nt] = __builtin_amdgcn_wmma_f32_16x16x32_bf16(
                false, aC[0].v, false, bC[nt].v, (short)0, acc[0][nt], false, false);
            acc[1][nt] = __builtin_amdgcn_wmma_f32_16x16x32_bf16(
                false, aC[1].v, false, bC[nt].v, (short)0, acc[1][nt], false, false);
        }
        const int kc = k0 + 64;
        if (kc < cC) {
            load_a(aC[0], aP0 + kc, half); load_a(aC[1], aP1 + kc, half);
#pragma unroll
            for (int nt = 0; nt < 4; ++nt) load_b(bC[nt], bP[nt] + kc);
        }
#pragma unroll
        for (int nt = 0; nt < 4; ++nt) {
            acc[0][nt] = __builtin_amdgcn_wmma_f32_16x16x32_bf16(
                false, aN[0].v, false, bN[nt].v, (short)0, acc[0][nt], false, false);
            acc[1][nt] = __builtin_amdgcn_wmma_f32_16x16x32_bf16(
                false, aN[1].v, false, bN[nt].v, (short)0, acc[1][nt], false, false);
        }
    }
#pragma unroll
    for (int mi = 0; mi < 2; ++mi) {
#pragma unroll
        for (int nt = 0; nt < 4; ++nt) {
#pragma unroll
            for (int r = 0; r < 8; ++r) {
                int n = n0 + nt * 16 + nIdx;
                int m = mrow + mi * 16 + r + 8 * half;
                out[(size_t)m * cC + n] = acc[mi][nt][r] + bo[n];
            }
        }
    }
}

// ---------------- launcher ----------------
extern "C" void kernel_launch(void* const* d_in, const int* in_sizes, int n_in,
                              void* d_out, int out_size, void* d_ws, size_t ws_size,
                              hipStream_t stream) {
    const float* x    = (const float*)d_in[0];
    const float* Wqkv = (const float*)d_in[1];
    const float* bqkv = (const float*)d_in[2];
    const float* Wo   = (const float*)d_in[3];
    const float* bo   = (const float*)d_in[4];
    float* out = (float*)d_out;

    char* ws = (char*)d_ws;
    constexpr size_t szX   = (size_t)cM * cC * 2;            // 16 MB
    constexpr size_t szWq  = (size_t)c3C * cC * 2;           // 6 MB
    constexpr size_t szWo  = (size_t)cC * cC * 2;            // 2 MB
    constexpr size_t szQKV = (size_t)cB * cH * cT * cD * 2;  // 16 MB each
    unsigned short* xbf    = (unsigned short*)(ws);
    unsigned short* wqkvT  = (unsigned short*)(ws + szX);
    unsigned short* woT    = (unsigned short*)(ws + szX + szWq);
    unsigned short* qbuf   = (unsigned short*)(ws + szX + szWq + szWo);
    unsigned short* kbuf   = (unsigned short*)(ws + szX + szWq + szWo + szQKV);
    unsigned short* vtbuf  = (unsigned short*)(ws + szX + szWq + szWo + 2 * szQKV);
    unsigned short* attnbf = (unsigned short*)(ws + szX + szWq + szWo + 3 * szQKV);

    int nX = cM * cC;
    k_cvt_bf16<<<(nX + 255) / 256, 256, 0, stream>>>(x, xbf, nX);
    k_transpose_bf16<<<(cC * c3C + 255) / 256, 256, 0, stream>>>(Wqkv, wqkvT, cC, c3C);
    k_transpose_bf16<<<(cC * cC + 255) / 256, 256, 0, stream>>>(Wo, woT, cC, cC);

    k_gemm_qkv<<<dim3(c3C / 64, cM / 256), 256, 0, stream>>>(xbf, wqkvT, bqkv, qbuf, kbuf, vtbuf);
    k_attn<<<dim3(cB * cH, cT / 128), 256, 0, stream>>>(qbuf, kbuf, vtbuf, attnbf);
    k_gemm_out<<<dim3(cC / 64, cM / 256), 256, 0, stream>>>(attnbf, woT, bo, out);
}
